// CausalSelfAttention_25838523252793
// MI455X (gfx1250) — matrix operand order repaired and checked by reading, not compiled, not run on hardware
//
#include <hip/hip_runtime.h>
#include <math.h>

// ---- CDNA5 WMMA types -------------------------------------------------------
typedef __attribute__((ext_vector_type(16))) __bf16 v16bf;
typedef __attribute__((ext_vector_type(8)))  __bf16 v8bf;
typedef __attribute__((ext_vector_type(8)))  float  v8f;
// ---- TDM descriptor vector types (6-arg builtin, clang-23 toolchain) -------
typedef __attribute__((ext_vector_type(4))) unsigned int u32x4;
typedef __attribute__((ext_vector_type(8))) int          i32x8;
typedef __attribute__((ext_vector_type(4))) int          i32x4;

__device__ __forceinline__ v16bf cat16(v8bf lo, v8bf hi) {
  return __builtin_shufflevector(lo, hi, 0,1,2,3,4,5,6,7,8,9,10,11,12,13,14,15);
}
__device__ __forceinline__ v8f wmma_bf16(v16bf a, v16bf b, v8f c) {
  // D(16x16,f32) = A(16x32,bf16) * B(32x16,bf16) + C
  return __builtin_amdgcn_wmma_f32_16x16x32_bf16(false, a, false, b, (short)0, c,
                                                 false, false);
}

// Issue a TDM 2D tile load: global (row-major, bf16) -> LDS, optional row pad.
// dims/strides in 2-byte elements. pad codes per D# spec:
//   pad_interval: 0=2,1=4,2=8,3=16,4=32,... DWORDs   pad_amount: code+1 DWORDs
__device__ __forceinline__ void tdm_load_2d(unsigned lds_off, const void* gptr,
                                            unsigned tile_d0, unsigned tile_d1,
                                            unsigned long long stride0,
                                            unsigned pad_int_code,
                                            unsigned pad_amt_code,
                                            unsigned pad_en) {
  unsigned long long ga = (unsigned long long)gptr;
  u32x4 g0;
  g0[0] = 1u;                                   // count=1, user descriptor
  g0[1] = lds_off;                              // lds_addr (bytes)
  g0[2] = (unsigned)(ga & 0xffffffffu);         // global_addr[31:0]
  g0[3] = (unsigned)((ga >> 32) & 0x1ffffffu) | (2u << 30);  // addr[56:32]|type=2
  unsigned flags = (1u << 16)                   // data_size = 2 bytes
                 | (pad_en << 20)               // pad_enable
                 | (pad_int_code << 22)
                 | (pad_amt_code << 25);
  i32x8 g1;
  g1[0] = (int)flags;                           // wg_mask=0 | size | pad ctl
  g1[1] = (int)((tile_d0 & 0xffffu) << 16);     // tensor_dim0[15:0] (== tile_d0)
  g1[2] = (int)((tile_d0 >> 16) | ((tile_d1 & 0xffffu) << 16)); // dim0 hi|dim1 lo
  g1[3] = (int)((tile_d1 >> 16) | (tile_d0 << 16));             // dim1 hi|tile_dim0
  g1[4] = (int)(tile_d1 & 0xffffu);             // tile_dim1 | tile_dim2=0
  g1[5] = (int)(stride0 & 0xffffffffu);         // tensor_dim0_stride[31:0]
  g1[6] = (int)((stride0 >> 32) & 0xffffu);     // stride0[47:32] | stride1 lo=0
  g1[7] = 0;
  i32x4 z4 = {0, 0, 0, 0};
  i32x8 z8 = {0, 0, 0, 0, 0, 0, 0, 0};
  __builtin_amdgcn_tensor_load_to_lds(g0, g1, z4, z4, z8, 0);
}
__device__ __forceinline__ unsigned lds_off_of(const void* p) {
  return (unsigned)(unsigned long long)p;       // low 32 bits = LDS byte offset
}

// Problem constants (B=4, T=2048, C=1024, H=16, HS=64)
#define TT 2048
#define CC 1024
#define HH 16
#define HS 64

// ---- precision conversion kernels ------------------------------------------
__global__ void convert_f32_bf16(const float* __restrict__ src,
                                 __bf16* __restrict__ dst, int n) {
  int i = blockIdx.x * blockDim.x + threadIdx.x;
  if (i < n) dst[i] = (__bf16)src[i];
}

// dst[n][k] = src[k][n]  (store weights N-major so GEMM B-frags are contiguous)
__global__ void transpose_f32_bf16(const float* __restrict__ src,
                                   __bf16* __restrict__ dst, int K, int N) {
  size_t i = (size_t)blockIdx.x * blockDim.x + threadIdx.x;
  int n = (int)(i / K), k = (int)(i % K);
  dst[i] = (__bf16)src[(size_t)k * N + n];
}

// ---- tiled bf16 GEMM: D[M,N] = A[M,K] * Bt[N,K]^T + bias -------------------
// TDM double-buffered tile staging; mode 0: scatter Q/K/V bf16, mode 1: fp32.
__global__ __launch_bounds__(256)
void gemm_bf16_kernel(const __bf16* __restrict__ A, const __bf16* __restrict__ Bt,
                      const float* __restrict__ bias, int K, int mode,
                      __bf16* __restrict__ Qo, __bf16* __restrict__ Ko,
                      __bf16* __restrict__ Vo, float* __restrict__ Fo, int ldf) {
  __shared__ __align__(16) __bf16 As[2][128][40];  // 128 x 32 (+8 pad via TDM)
  __shared__ __align__(16) __bf16 Bs[2][128][40];
  const int tid = threadIdx.x;
  const int lane = tid & 31, w = tid >> 5;
  const int lm = lane & 15, lh = lane >> 4;
  const int wm = w & 1, wn = w >> 1;               // 2 waves in M, 4 in N
  const int rowBase = blockIdx.y * 128;
  const int colBase = blockIdx.x * 128;

  const v8f vzero = {0.f,0.f,0.f,0.f,0.f,0.f,0.f,0.f};
  v8f acc[4][2];
  for (int i = 0; i < 4; ++i)
    for (int j = 0; j < 2; ++j) acc[i][j] = vzero;

  const int nk = K / 32;
  // one wave drives the Tensor Data Mover; pad: 16-DWORD rows + 4-DWORD pad
  if (w == 0) {
    tdm_load_2d(lds_off_of(&As[0][0][0]), A + (size_t)rowBase * K, 32, 128,
                (unsigned long long)K, 3u, 3u, 1u);
    tdm_load_2d(lds_off_of(&Bs[0][0][0]), Bt + (size_t)colBase * K, 32, 128,
                (unsigned long long)K, 3u, 3u, 1u);
  }
  for (int kt = 0; kt < nk; ++kt) {
    const int cur = kt & 1;
    if (w == 0) {
      if (kt + 1 < nk) {                           // prefetch next K-tile
        const int kB = (kt + 1) * 32;
        tdm_load_2d(lds_off_of(&As[cur ^ 1][0][0]),
                    A + (size_t)rowBase * K + kB, 32, 128,
                    (unsigned long long)K, 3u, 3u, 1u);
        tdm_load_2d(lds_off_of(&Bs[cur ^ 1][0][0]),
                    Bt + (size_t)colBase * K + kB, 32, 128,
                    (unsigned long long)K, 3u, 3u, 1u);
        __builtin_amdgcn_s_wait_tensorcnt(2);      // current pair landed
      } else {
        __builtin_amdgcn_s_wait_tensorcnt(0);
      }
    }
    __syncthreads();
    v16bf af[4], bf2[2];
    for (int i = 0; i < 4; ++i) {
      int r = wm * 64 + i * 16 + lm;               // A-layout: lane<16 K0-7/16-23
      af[i] = cat16(*(const v8bf*)&As[cur][r][lh * 8],
                    *(const v8bf*)&As[cur][r][16 + lh * 8]);
    }
    for (int j = 0; j < 2; ++j) {
      int r = wn * 32 + j * 16 + lm;               // B-layout: lane<16 K0-7/16-23, N=lm
      bf2[j] = cat16(*(const v8bf*)&Bs[cur][r][lh * 8],
                     *(const v8bf*)&Bs[cur][r][16 + lh * 8]);
    }
    for (int i = 0; i < 4; ++i)
      for (int j = 0; j < 2; ++j) acc[i][j] = wmma_bf16(af[i], bf2[j], acc[i][j]);
    __syncthreads();                               // free buffer for kt+2 DMA
  }

  for (int i = 0; i < 4; ++i)
    for (int j = 0; j < 2; ++j) {
      int col = colBase + wn * 32 + j * 16 + lm;
      float bv = bias[col];
      for (int r = 0; r < 8; ++r) {
        int row = rowBase + wm * 64 + i * 16 + r + lh * 8;  // C/D: M=r / 8+r
        float vl = acc[i][j][r] + bv;
        if (mode == 0) {
          int sect = col >> 10, ci = col & 1023, head = ci >> 6, d = ci & 63;
          if (sect == 0) vl *= 0.125f;             // fold 1/sqrt(HS) into Q
          __bf16* dst = (sect == 0) ? Qo : (sect == 1 ? Ko : Vo);
          int bb = row >> 11, t = row & (TT - 1);
          dst[((((size_t)bb * HH + head) * TT + t) << 6) + d] = (__bf16)vl;
        } else {
          Fo[(size_t)row * ldf + col] = vl;
        }
      }
    }
}

// ---- flash attention: one (b,h), 128 queries per workgroup -----------------
__global__ __launch_bounds__(256)
void attn_kernel(const __bf16* __restrict__ Q, const __bf16* __restrict__ Kt,
                 const __bf16* __restrict__ V, __bf16* __restrict__ Y) {
  __shared__ __align__(16) __bf16 Qs[128][72];   // 128 q x 64 d
  __shared__ __align__(16) __bf16 Ks[32][72];    // 32 k  x 64 d (TDM-padded)
  __shared__ __align__(16) __bf16 Vts[64][40];   // transposed: 64 d x 32 k
  __shared__ __align__(16) __bf16 Ps[8][16][40]; // per-wave P scratch 16x32
  const int tid = threadIdx.x;
  const int lane = tid & 31, w = tid >> 5;
  const int lm = lane & 15, lh = lane >> 4;
  const int bh = blockIdx.y;
  const int qbase = blockIdx.x * 128;
  const __bf16* Qh = Q  + (size_t)bh * TT * HS;
  const __bf16* Kh = Kt + (size_t)bh * TT * HS;
  const __bf16* Vh = V  + (size_t)bh * TT * HS;

  for (int c = tid; c < 1024; c += 256) {        // Q tile: 1024 16B chunks
    int r = c >> 3, c8 = (c & 7) * 8;
    *(v8bf*)&Qs[r][c8] = *(const v8bf*)&Qh[(size_t)(qbase + r) * HS + c8];
  }
  __syncthreads();
  v16bf aq[2];
  {
    int r = w * 16 + lm;
    aq[0] = cat16(*(const v8bf*)&Qs[r][lh * 8],      *(const v8bf*)&Qs[r][16 + lh * 8]);
    aq[1] = cat16(*(const v8bf*)&Qs[r][32 + lh * 8], *(const v8bf*)&Qs[r][48 + lh * 8]);
  }
  float rmax[8], rsum[8];
  for (int r = 0; r < 8; ++r) { rmax[r] = -3.0e38f; rsum[r] = 0.f; }
  const v8f vzero = {0.f,0.f,0.f,0.f,0.f,0.f,0.f,0.f};
  v8f o[4];
  for (int g = 0; g < 4; ++g) o[g] = vzero;

  const int nkb = qbase / 32 + 4;                // cover keys 0 .. qbase+127
  for (int kb = 0; kb < nkb; ++kb) {
    const int k0 = kb * 32;
    __syncthreads();                             // done reading previous tiles
    // K tile via TDM (32 rows x 64 el; pad 32-DWORD rows by 4 DWORDs -> stride 72)
    if (w == 0)
      tdm_load_2d(lds_off_of(&Ks[0][0]), Kh + (size_t)k0 * HS, 64, 32,
                  (unsigned long long)HS, 4u, 3u, 1u);
    {                                            // V tile: load + transpose
      int r = tid >> 3, c8 = (tid & 7) * 8;
      v8bf vv = *(const v8bf*)&Vh[(size_t)(k0 + r) * HS + c8];
      for (int jj = 0; jj < 8; ++jj) Vts[c8 + jj][r] = vv[jj];
    }
    if (w == 0) __builtin_amdgcn_s_wait_tensorcnt(0);
    __syncthreads();

    // S = Q * K^T  (two 16-key subtiles, each K=64 via two chained WMMAs)
    int r0 = lm, r1 = 16 + lm;
    v16bf bk00 = cat16(*(const v8bf*)&Ks[r0][lh * 8],      *(const v8bf*)&Ks[r0][16 + lh * 8]);
    v16bf bk01 = cat16(*(const v8bf*)&Ks[r0][32 + lh * 8], *(const v8bf*)&Ks[r0][48 + lh * 8]);
    v16bf bk10 = cat16(*(const v8bf*)&Ks[r1][lh * 8],      *(const v8bf*)&Ks[r1][16 + lh * 8]);
    v16bf bk11 = cat16(*(const v8bf*)&Ks[r1][32 + lh * 8], *(const v8bf*)&Ks[r1][48 + lh * 8]);
    v8f s0 = wmma_bf16(aq[1], bk01, wmma_bf16(aq[0], bk00, vzero));
    v8f s1 = wmma_bf16(aq[1], bk11, wmma_bf16(aq[0], bk10, vzero));

    // online softmax (row = r + 8*lh across half-waves, col = lm)
    const int qg = qbase + w * 16 + lh * 8;
    const int kg0 = k0 + lm, kg1 = k0 + 16 + lm;
    float alpha[8];
    for (int r = 0; r < 8; ++r) {
      float e0 = (kg0 <= qg + r) ? s0[r] : -3.0e38f;   // causal mask
      float e1 = (kg1 <= qg + r) ? s1[r] : -3.0e38f;
      float mv = fmaxf(e0, e1);
      mv = fmaxf(mv, __shfl_xor(mv, 1, 32));
      mv = fmaxf(mv, __shfl_xor(mv, 2, 32));
      mv = fmaxf(mv, __shfl_xor(mv, 4, 32));
      mv = fmaxf(mv, __shfl_xor(mv, 8, 32));
      float mnew = fmaxf(rmax[r], mv);
      float a = __expf(rmax[r] - mnew);
      rmax[r] = mnew;
      float p0 = __expf(e0 - mnew);
      float p1 = __expf(e1 - mnew);
      float ps = p0 + p1;
      ps += __shfl_xor(ps, 1, 32);
      ps += __shfl_xor(ps, 2, 32);
      ps += __shfl_xor(ps, 4, 32);
      ps += __shfl_xor(ps, 8, 32);
      rsum[r] = rsum[r] * a + ps;
      alpha[r] = a;
      int ml = r + lh * 8;
      Ps[w][ml][lm]      = (__bf16)p0;   // C/D layout -> LDS
      Ps[w][ml][16 + lm] = (__bf16)p1;
    }
    for (int g = 0; g < 4; ++g)
      for (int r = 0; r < 8; ++r) o[g][r] *= alpha[r];

    asm volatile("s_wait_dscnt 0" ::: "memory");       // P store -> A-frag load

    v16bf pf = cat16(*(const v8bf*)&Ps[w][lm][lh * 8],
                     *(const v8bf*)&Ps[w][lm][16 + lh * 8]);
    for (int g = 0; g < 4; ++g) {
      int d = g * 16 + lm;
      v16bf bv = cat16(*(const v8bf*)&Vts[d][lh * 8],
                       *(const v8bf*)&Vts[d][16 + lh * 8]);
      o[g] = wmma_bf16(pf, bv, o[g]);                  // O += P * V
    }
  }

  // epilogue: normalize, write [B,T,C] bf16
  const int b = bh >> 4, h = bh & 15;
  for (int r = 0; r < 8; ++r) {
    float inv = 1.0f / rsum[r];
    int t = qbase + w * 16 + r + lh * 8;
    for (int g = 0; g < 4; ++g) {
      float vl = o[g][r] * inv;
      Y[(((size_t)b * TT + t) << 10) + h * HS + g * 16 + lm] = (__bf16)vl;
    }
  }
}

// ---- launcher --------------------------------------------------------------
extern "C" void kernel_launch(void* const* d_in, const int* in_sizes, int n_in,
                              void* d_out, int out_size, void* d_ws, size_t ws_size,
                              hipStream_t stream) {
  const float* x      = (const float*)d_in[0];   // [4,2048,1024]
  const float* w_attn = (const float*)d_in[1];   // [1024,3072]
  const float* b_attn = (const float*)d_in[2];   // [3072]
  const float* w_proj = (const float*)d_in[3];   // [1024,1024]
  const float* b_proj = (const float*)d_in[4];   // [1024]
  float* out = (float*)d_out;                    // [4,2048,1024] fp32

  char* ws = (char*)d_ws;                        // needs 88 MB total
  __bf16* Xbf = (__bf16*)(ws);                   // 16 MB  x in bf16
  __bf16* WaT = (__bf16*)(ws + (16u << 20));     //  6 MB  w_attn^T bf16
  __bf16* WpT = (__bf16*)(ws + (22u << 20));     //  2 MB  w_proj^T bf16
  __bf16* Qb  = (__bf16*)(ws + (24u << 20));     // 16 MB  Q [B,H,T,HS]
  __bf16* Kb  = (__bf16*)(ws + (40u << 20));     // 16 MB  K
  __bf16* Vb  = (__bf16*)(ws + (56u << 20));     // 16 MB  V
  __bf16* Yb  = (__bf16*)(ws + (72u << 20));     // 16 MB  attn out [B,T,C]

  const int NX = 4 * TT * CC;                    // 8388608
  convert_f32_bf16<<<NX / 256, 256, 0, stream>>>(x, Xbf, NX);
  transpose_f32_bf16<<<(3 * CC * CC) / 256, 256, 0, stream>>>(w_attn, WaT, CC, 3 * CC);
  transpose_f32_bf16<<<(CC * CC) / 256, 256, 0, stream>>>(w_proj, WpT, CC, CC);

  // QKV = X * Wattn + b  -> scatter bf16 Q/K/V (Q pre-scaled)
  gemm_bf16_kernel<<<dim3(24, 64), 256, 0, stream>>>(
      Xbf, WaT, b_attn, CC, /*mode=*/0, Qb, Kb, Vb, nullptr, 0);

  // flash attention per (b,h) / 128-query block
  attn_kernel<<<dim3(TT / 128, 4 * HH), 256, 0, stream>>>(Qb, Kb, Vb, Yb);

  // out = Y * Wproj + b
  gemm_bf16_kernel<<<dim3(8, 64), 256, 0, stream>>>(
      Yb, WpT, b_proj, CC, /*mode=*/1, nullptr, nullptr, nullptr, out, CC);
}
